// AttentionModel_39848706573351
// MI455X (gfx1250) — compile-verified, matrix operand order checked by reading, not run on hardware
//
#include <hip/hip_runtime.h>

typedef __attribute__((ext_vector_type(16))) _Float16 v16h;
typedef __attribute__((ext_vector_type(8)))  _Float16 v8h;
typedef __attribute__((ext_vector_type(8)))  float    v8f;

#define B_   4
#define S_   2048
#define E_   1024
#define TQ   16            // query rows per workgroup
#define TK   128           // keys per iteration (16 per wave)
#define NW   8             // waves per workgroup (wave32)
#define CPW  (E_ / NW)     // 128 output columns per wave
#define QS_LD (E_ + 8)     // Q tile row stride (halfs), bank-conflict pad
#define P_LD  (TK + 8)     // P tile row stride (halfs), bank-conflict pad

// ---- fragment builders (ISA 7.12.2 16-bit 16x32 layout) ----
// A/B fragment from a row of f16 (LDS or global): 2x b128 loads.
__device__ __forceinline__ v16h frag_from_f16row(const _Float16* p, int off8) {
    v8h lo = *(const v8h*)(p + off8);
    v8h hi = *(const v8h*)(p + 16 + off8);
    v16h f;
#pragma unroll
    for (int i = 0; i < 8; ++i) { f[i] = lo[i]; f[i + 8] = hi[i]; }
    return f;
}

// Fragment from a row of f32 in global, converted in flight (fallback path).
__device__ __forceinline__ v16h frag_from_f32row(const float* __restrict__ p, int off8) {
    const float4* a0 = (const float4*)(p + off8);
    const float4* a1 = (const float4*)(p + 16 + off8);
    float4 x0 = a0[0], x1 = a0[1], y0 = a1[0], y1 = a1[1];
    v16h f;
    f[0]  = (_Float16)x0.x; f[1]  = (_Float16)x0.y;
    f[2]  = (_Float16)x0.z; f[3]  = (_Float16)x0.w;
    f[4]  = (_Float16)x1.x; f[5]  = (_Float16)x1.y;
    f[6]  = (_Float16)x1.z; f[7]  = (_Float16)x1.w;
    f[8]  = (_Float16)y0.x; f[9]  = (_Float16)y0.y;
    f[10] = (_Float16)y0.z; f[11] = (_Float16)y0.w;
    f[12] = (_Float16)y1.x; f[13] = (_Float16)y1.y;
    f[14] = (_Float16)y1.z; f[15] = (_Float16)y1.w;
    return f;
}

// B fragment gathered down a column of f32 in global (fallback path).
__device__ __forceinline__ v16h frag_from_f32col(const float* __restrict__ p, int off8) {
    v16h f;
#pragma unroll
    for (int i = 0; i < 8; ++i) {
        f[i]     = (_Float16)p[(size_t)(off8 + i) * E_];
        f[i + 8] = (_Float16)p[(size_t)(16 + off8 + i) * E_];
    }
    return f;
}

// ---- prep kernels (one memory-bound sweep, ~2us at 23.3 TB/s) ----
__global__ __launch_bounds__(256)
void convert_f16(const float* __restrict__ src, _Float16* __restrict__ dst) {
    size_t i = ((size_t)blockIdx.x * 256 + threadIdx.x) * 8;
    float4 a = *(const float4*)(src + i);
    float4 b = *(const float4*)(src + i + 4);
    v8h h;
    h[0] = (_Float16)a.x; h[1] = (_Float16)a.y; h[2] = (_Float16)a.z; h[3] = (_Float16)a.w;
    h[4] = (_Float16)b.x; h[5] = (_Float16)b.y; h[6] = (_Float16)b.z; h[7] = (_Float16)b.w;
    *(v8h*)(dst + i) = h;
}

// V (B,S,E) f32  ->  Vt (B,E,S) f16, LDS-tiled transpose.
__global__ __launch_bounds__(256)
void transpose_f16(const float* __restrict__ src, _Float16* __restrict__ dst) {
    __shared__ _Float16 t[32][33];
    const int b  = blockIdx.z;
    const int e0 = blockIdx.x * 32;
    const int s0 = blockIdx.y * 32;
    const float* Sg = src + (size_t)b * S_ * E_;
    _Float16*    Dg = dst + (size_t)b * E_ * S_;
    for (int r = threadIdx.y; r < 32; r += 8)
        t[r][threadIdx.x] = (_Float16)Sg[(size_t)(s0 + r) * E_ + e0 + threadIdx.x];
    __syncthreads();
    for (int r = threadIdx.y; r < 32; r += 8)
        Dg[(size_t)(e0 + r) * S_ + s0 + threadIdx.x] = t[threadIdx.x][r];
}

// ---- flash attention main kernel ----
template <bool FAST>
__global__ __launch_bounds__(256)
void attn_fwd(const float* __restrict__ Q, const void* __restrict__ Kp,
              const void* __restrict__ Vp, const float* __restrict__ M,
              float* __restrict__ O)
{
    __shared__ __align__(16) _Float16 Qs[TQ][QS_LD];  // pre-scaled f16 Q tile
    __shared__ __align__(16) float    Sp[TQ][TK];     // fp32 scores 16x128
    __shared__ __align__(16) _Float16 Ps[TQ][P_LD];   // f16 probabilities
    __shared__ __align__(16) float    red[TQ][16];    // softmax partials
    __shared__ float m_s[TQ], l_s[TQ], a_s[TQ];

    const int tid  = threadIdx.x;
    const int lane = tid & 31;
    const int w    = tid >> 5;
    const int q0   = blockIdx.x * TQ;
    const int b    = blockIdx.y;

    const float* Qg = Q + (size_t)b * S_ * E_ + (size_t)q0 * E_;
    const float* Mg = M + (size_t)b * S_ * S_ + (size_t)q0 * S_;
    const _Float16* Kh = (const _Float16*)Kp + (size_t)b * S_ * E_;   // FAST
    const _Float16* Vt = (const _Float16*)Vp + (size_t)b * E_ * S_;   // FAST (E,S)
    const float* Kf = (const float*)Kp + (size_t)b * S_ * E_;         // fallback
    const float* Vf = (const float*)Vp + (size_t)b * S_ * E_;         // fallback

    // stage Q tile to LDS as f16, 1/sqrt(1024)=1/32 folded in
    for (int i = tid; i < TQ * E_; i += 256) {
        int r = i >> 10, c = i & (E_ - 1);
        Qs[r][c] = (_Float16)(Qg[(size_t)r * E_ + c] * 0.03125f);
    }
    if (tid < TQ) { m_s[tid] = -INFINITY; l_s[tid] = 0.0f; }

    v8f Oacc[8];
#pragma unroll
    for (int nt = 0; nt < 8; ++nt) Oacc[nt] = (v8f){0,0,0,0,0,0,0,0};

    const int off8 = (lane < 16) ? 0 : 8;
    const int mrow = lane & 15;
    const int cm   = (lane < 16) ? 0 : 8;
    const int srow = tid >> 4;
    const int sch  = tid & 15;

    for (int j = 0; j < S_ / TK; ++j) {
        // ---- GEMM1: S = Q.K^T, full-E reduction, 16 keys per wave ----
        const int krowi = j * TK + w * 16 + mrow;
        if (j + 1 < S_ / TK) {   // global_prefetch_b8 of next K row
            if constexpr (FAST) {
                __builtin_prefetch(Kh + (size_t)(krowi + TK) * E_, 0, 3);
                __builtin_prefetch(Kh + (size_t)(krowi + TK) * E_ + 512, 0, 3);
            } else {
                __builtin_prefetch(Kf + (size_t)(krowi + TK) * E_, 0, 3);
                __builtin_prefetch(Kf + (size_t)(krowi + TK) * E_ + 512, 0, 3);
            }
        }
        v8f sacc = (v8f){0,0,0,0,0,0,0,0};
#pragma unroll 8
        for (int ec = 0; ec < E_ / 32; ++ec) {
            const int e0 = ec * 32;
            v16h af = frag_from_f16row(&Qs[mrow][e0], off8);
            v16h bf;
            if constexpr (FAST) bf = frag_from_f16row(Kh + (size_t)krowi * E_ + e0, off8);
            else                bf = frag_from_f32row(Kf + (size_t)krowi * E_ + e0, off8);
            sacc = __builtin_amdgcn_wmma_f32_16x16x32_f16(
                false, af, false, bf, (short)0, sacc, false, false);
        }
#pragma unroll
        for (int r = 0; r < 8; ++r)
            Sp[r + cm][w * 16 + mrow] = sacc[r];   // disjoint slice, no atomics
        __syncthreads();

        // ---- online softmax over 128 keys: 16 threads/row x 8 keys each ----
        float sv[8];
        float lmax = -INFINITY;
#pragma unroll
        for (int i = 0; i < 8; ++i) {
            const int kk = sch * 8 + i;
            float s = Sp[srow][kk] + Mg[(size_t)srow * S_ + j * TK + kk];
            sv[i] = s;
            lmax = fmaxf(lmax, s);
        }
        red[srow][sch] = lmax;
        __syncthreads();
        if (tid < TQ) {
            float mold = m_s[tid], mnew = mold;
#pragma unroll
            for (int i = 0; i < 16; ++i) mnew = fmaxf(mnew, red[tid][i]);
            a_s[tid] = __expf(mold - mnew);
            m_s[tid] = mnew;
        }
        __syncthreads();
        const float mnew = m_s[srow];
        float lsum = 0.0f;
#pragma unroll
        for (int i = 0; i < 8; ++i) {
            float p = __expf(sv[i] - mnew);
            lsum += p;
            Ps[srow][sch * 8 + i] = (_Float16)p;
        }
        red[srow][sch] = lsum;
        __syncthreads();
        if (tid < TQ) {
            float t = 0.0f;
#pragma unroll
            for (int i = 0; i < 16; ++i) t += red[tid][i];
            l_s[tid] = l_s[tid] * a_s[tid] + t;
        }
        __syncthreads();

        // ---- GEMM2: O = alpha*O + P(16x128).V(128x128 cols/wave) ----
        float aR[8];
#pragma unroll
        for (int r = 0; r < 8; ++r) aR[r] = a_s[r + cm];

        v16h pfr[4];
#pragma unroll
        for (int kc = 0; kc < 4; ++kc)
            pfr[kc] = frag_from_f16row(&Ps[mrow][kc * 32], off8);

#pragma unroll
        for (int nt = 0; nt < 8; ++nt) {
#pragma unroll
            for (int r = 0; r < 8; ++r) Oacc[nt][r] *= aR[r];
            const int col = w * CPW + nt * 16 + mrow;
#pragma unroll
            for (int kc = 0; kc < 4; ++kc) {
                v16h vf;
                if constexpr (FAST)   // Vt row read: contiguous keys, no gather
                    vf = frag_from_f16row(Vt + (size_t)col * S_ + j * TK + kc * 32, off8);
                else
                    vf = frag_from_f32col(Vf + (size_t)(j * TK + kc * 32) * E_ + col, off8);
                Oacc[nt] = __builtin_amdgcn_wmma_f32_16x16x32_f16(
                    false, pfr[kc], false, vf, (short)0, Oacc[nt], false, false);
            }
        }
        __syncthreads();
    }

    // ---- normalize and store ----
    float linv[8];
#pragma unroll
    for (int r = 0; r < 8; ++r) linv[r] = 1.0f / l_s[r + cm];

    float* Orow = O + (size_t)b * S_ * E_ + (size_t)q0 * E_;
#pragma unroll
    for (int nt = 0; nt < 8; ++nt) {
        const int col = w * CPW + nt * 16 + mrow;
#pragma unroll
        for (int r = 0; r < 8; ++r)
            Orow[(size_t)(r + cm) * E_ + col] = Oacc[nt][r] * linv[r];
    }
}

extern "C" void kernel_launch(void* const* d_in, const int* in_sizes, int n_in,
                              void* d_out, int out_size, void* d_ws, size_t ws_size,
                              hipStream_t stream) {
    (void)in_sizes; (void)n_in; (void)out_size;
    const float* q = (const float*)d_in[0];
    const float* k = (const float*)d_in[1];
    const float* v = (const float*)d_in[2];
    const float* m = (const float*)d_in[3];
    float* out = (float*)d_out;

    const size_t elems = (size_t)B_ * S_ * E_;
    const size_t need  = elems * sizeof(_Float16) * 2;   // Kh + Vt = 32 MB
    dim3 agrid(S_ / TQ, B_);

    if (d_ws && ws_size >= need) {
        _Float16* Kh = (_Float16*)d_ws;
        _Float16* Vt = Kh + elems;
        convert_f16<<<(int)(elems / (256 * 8)), 256, 0, stream>>>(k, Kh);
        transpose_f16<<<dim3(E_ / 32, S_ / 32, B_), dim3(32, 8), 0, stream>>>(v, Vt);
        attn_fwd<true><<<agrid, 256, 0, stream>>>(q, Kh, Vt, m, out);
    } else {
        attn_fwd<false><<<agrid, 256, 0, stream>>>(q, k, v, m, out);
    }
}
